// GRUConv3d_3478923509775
// MI455X (gfx1250) — matrix operand order repaired and checked by reading, not compiled, not run on hardware
//
#include <hip/hip_runtime.h>
#include <math.h>

typedef __attribute__((ext_vector_type(16))) _Float16 v16h;
typedef __attribute__((ext_vector_type(8)))  _Float16 v8h;
typedef __attribute__((ext_vector_type(8)))  float    v8f;

#define D_ 32
#define DP 34              // padded dim
#define CIN 32
#define COUT 32
#define PIX_PAD (DP*DP*DP) // 39304 per batch
#define NPIX (D_*D_*D_)    // 32768 per batch
#define TAPS 27
// wprep: [dir][conv][tap][ntile][lane(32)][slot(16)] f16
#define WTILE_ELEMS 512
#define CONV_STRIDE (TAPS*2*WTILE_ELEMS)   // 27648 elems between convs

__device__ __forceinline__ float sigmoidf_(float x) {
    return 1.0f / (1.0f + expf(-x));
}

// ---------------------------------------------------------------- prep x: NCDHW f32 -> padded NDHWC f16
__global__ void prep_x(const float* __restrict__ x, _Float16* __restrict__ xpad, int n) {
    int tid = blockIdx.x * blockDim.x + threadIdx.x;
    if (tid >= n) return;
    int ci = tid & 31;
    int t  = tid >> 5;
    int wz = t % DP; t /= DP;
    int hz = t % DP; t /= DP;
    int dz = t % DP; t /= DP;
    int b  = t;
    int d = dz - 1, h = hz - 1, w = wz - 1;
    float v = 0.0f;
    if ((unsigned)d < D_ && (unsigned)h < D_ && (unsigned)w < D_)
        v = x[((size_t)(b * CIN + ci) * D_ + d) * (D_ * D_) + h * D_ + w];
    xpad[tid] = (_Float16)v;
}

// ---------------------------------------------------------------- prep weights into per-lane WMMA B layout
__global__ void prep_w(const float* __restrict__ Wh, const float* __restrict__ Wz,
                       const float* __restrict__ Ws, _Float16* __restrict__ wprep, int n) {
    int tid = blockIdx.x * blockDim.x + threadIdx.x;
    if (tid >= n) return;
    int slot = tid & 15;          // K = 16*(slot/8) + 8*(lane/16) + slot%8
    int t = tid >> 4;
    int lane = t & 31;  t >>= 5;  // N + 16*((K/8)%2)
    int nt   = t & 1;   t >>= 1;
    int tap  = t % TAPS; t /= TAPS;
    int conv = t % 3;    t /= 3;
    int dir  = t;
    int k_ci = (slot & 8) * 2 + (lane >> 4) * 8 + (slot & 7);   // input channel
    int co   = nt * 16 + (lane & 15);         // output channel
    const float* W = (conv == 0) ? Wh : (conv == 1) ? Wz : Ws;
    float v = W[((size_t)(dir * COUT + co) * CIN + k_ci) * TAPS + tap];
    wprep[tid] = (_Float16)v;
}

// ---------------------------------------------------------------- zero workspace accumulator
__global__ void zero_f32(float* __restrict__ p, int n) {
    int tid = blockIdx.x * blockDim.x + threadIdx.x;
    if (tid < n) p[tid] = 0.0f;
}

// ---------------------------------------------------------------- implicit-GEMM conv3d, 3 convs fused, WMMA
__global__ __launch_bounds__(256) void conv_wmma(
    const _Float16* __restrict__ xpad, const _Float16* __restrict__ wprep,
    const float* __restrict__ bh, const float* __restrict__ bz, const float* __restrict__ bs,
    float* __restrict__ hbuf, float* __restrict__ zbuf, float* __restrict__ sbuf, int dir)
{
    const int wave = threadIdx.x >> 5;
    const int lane = threadIdx.x & 31;
    const int tIdx = blockIdx.x * 8 + wave;   // 0..4095
    const int w0 = (tIdx & 1) * 16;
    const int h  = (tIdx >> 1) & 31;
    const int d  = (tIdx >> 6) & 31;
    const int b  =  tIdx >> 11;
    const int m    = lane & 15;   // A row (pixel along W)
    const int half = lane >> 4;   // K-half selector

    v8f ch0 = {}, ch1 = {}, cz0 = {}, cz1 = {}, cs0 = {}, cs1 = {};

    const _Float16* wdir = wprep + (size_t)dir * 3 * CONV_STRIDE;

    for (int kd = 0; kd < 3; ++kd)
    for (int kh = 0; kh < 3; ++kh)
    for (int kw = 0; kw < 3; ++kw) {
        const int tap = (kd * 3 + kh) * 3 + kw;
        // A operand: 16 pixels x 32 Cin from padded NDHWC (ci contiguous).
        const size_t pix = ((size_t)(b * DP + d + kd) * DP + (h + kh)) * DP + (w0 + m + kw);
        const v8h* ap = (const v8h*)(xpad + pix * CIN + half * 8);
        v8h lo = ap[0];   // ci [8*half, 8*half+8)
        v8h hi = ap[2];   // ci [16+8*half, ...)
        v16h a = __builtin_shufflevector(lo, hi, 0,1,2,3,4,5,6,7,8,9,10,11,12,13,14,15);

        const _Float16* wl = wdir + (size_t)tap * 2 * WTILE_ELEMS + lane * 16;
        v16h wh0 = *(const v16h*)(wl);
        v16h wh1 = *(const v16h*)(wl + WTILE_ELEMS);
        v16h wz0 = *(const v16h*)(wl + CONV_STRIDE);
        v16h wz1 = *(const v16h*)(wl + CONV_STRIDE + WTILE_ELEMS);
        v16h ws0 = *(const v16h*)(wl + 2 * CONV_STRIDE);
        v16h ws1 = *(const v16h*)(wl + 2 * CONV_STRIDE + WTILE_ELEMS);

        ch0 = __builtin_amdgcn_wmma_f32_16x16x32_f16(false, a, false, wh0, (short)0, ch0, false, false);
        ch1 = __builtin_amdgcn_wmma_f32_16x16x32_f16(false, a, false, wh1, (short)0, ch1, false, false);
        cz0 = __builtin_amdgcn_wmma_f32_16x16x32_f16(false, a, false, wz0, (short)0, cz0, false, false);
        cz1 = __builtin_amdgcn_wmma_f32_16x16x32_f16(false, a, false, wz1, (short)0, cz1, false, false);
        cs0 = __builtin_amdgcn_wmma_f32_16x16x32_f16(false, a, false, ws0, (short)0, cs0, false, false);
        cs1 = __builtin_amdgcn_wmma_f32_16x16x32_f16(false, a, false, ws1, (short)0, cs1, false, false);
    }

    // Epilogue: bias + sigmoid(z,s); store NDHWC fp32 (lanes 0..15 -> consecutive co).
    const int co = lane & 15;
    const float bH0 = bh[dir * COUT + co],      bH1 = bh[dir * COUT + co + 16];
    const float bZ0 = bz[dir * COUT + co],      bZ1 = bz[dir * COUT + co + 16];
    const float bS0 = bs[dir * COUT + co],      bS1 = bs[dir * COUT + co + 16];
    #pragma unroll
    for (int v = 0; v < 8; ++v) {
        const int M = v + half * 8;   // pixel row of this VGPR
        const size_t o = ((size_t)((b * D_ + d) * D_ + h) * D_ + (w0 + M)) * COUT;
        hbuf[o + co]      = ch0[v] + bH0;
        hbuf[o + co + 16] = ch1[v] + bH1;
        zbuf[o + co]      = sigmoidf_(cz0[v] + bZ0);
        zbuf[o + co + 16] = sigmoidf_(cz1[v] + bZ1);
        sbuf[o + co]      = sigmoidf_(cs0[v] + bS0);
        sbuf[o + co + 16] = sigmoidf_(cs1[v] + bS1);
    }
}

// ---------------------------------------------------------------- diagonal GRU scan, fused out += h*s
// One wave per diagonal chain; lane = channel (coalesced over channel-contiguous NDHWC buffers).
__global__ void gru_scan(const float* __restrict__ zbuf, const float* __restrict__ hbuf,
                         const float* __restrict__ sbuf, const float* __restrict__ h0,
                         float* __restrict__ outws, int dir)
{
    const int lane = threadIdx.x & 31;                               // channel
    const int grp  = (blockIdx.x * blockDim.x + threadIdx.x) >> 5;   // 0..65535
    const int cell = grp & (NPIX - 1);
    const int b    = grp >> 15;
    int k = cell & 31, j = (cell >> 5) & 31, i = cell >> 10;
    const int di = ((dir >> 2) & 1) ? 1 : -1;
    const int dj = ((dir >> 1) & 1) ? 1 : -1;
    const int dk = ( dir       & 1) ? 1 : -1;
    const int fi = (di > 0) ? i : 31 - i;
    const int fj = (dj > 0) ? j : 31 - j;
    const int fk = (dk > 0) ? k : 31 - k;
    if (fi != 0 && fj != 0 && fk != 0) return;   // not a chain start (wave-uniform)
    int mx = fi > fj ? fi : fj; if (fk > mx) mx = fk;
    const int steps = D_ - mx;

    float hv = h0[dir * COUT + lane];
    for (int t = 0; t < steps; ++t) {
        const size_t idx = ((size_t)((b * D_ + i) * D_ + j) * D_ + k) * COUT + lane;
        const float z = zbuf[idx];
        hv = z * hbuf[idx] + (1.0f - z) * hv;
        outws[idx] += hv * sbuf[idx];            // chains partition cells: race-free per dir
        i += di; j += dj; k += dk;
    }
}

// ---------------------------------------------------------------- NDHWC -> NCDHW final transpose
__global__ void to_ncdhw(const float* __restrict__ outws, float* __restrict__ out, int n) {
    int tid = blockIdx.x * blockDim.x + threadIdx.x;
    if (tid >= n) return;
    const int pix = tid & (NPIX - 1);
    const int c   = (tid >> 15) & 31;
    const int b   =  tid >> 20;
    out[tid] = outws[((size_t)b * NPIX + pix) * COUT + c];
}

extern "C" void kernel_launch(void* const* d_in, const int* in_sizes, int n_in,
                              void* d_out, int out_size, void* d_ws, size_t ws_size,
                              hipStream_t stream) {
    const float* x  = (const float*)d_in[0];
    const float* Wh = (const float*)d_in[1];
    const float* bh = (const float*)d_in[2];
    const float* Wz = (const float*)d_in[3];
    const float* bz = (const float*)d_in[4];
    const float* Ws = (const float*)d_in[5];
    const float* bs = (const float*)d_in[6];
    const float* h0 = (const float*)d_in[7];
    float* out = (float*)d_out;

    char* ws = (char*)d_ws;
    const size_t xpad_bytes = (size_t)2 * PIX_PAD * CIN * sizeof(_Float16);   // 5,030,912
    const size_t wprep_n    = (size_t)8 * 3 * TAPS * 2 * WTILE_ELEMS;         // 663,552 elems
    const size_t wprep_bytes = wprep_n * sizeof(_Float16);                    // 1,327,104
    const size_t buf_elems  = (size_t)2 * NPIX * COUT;                        // 2,097,152
    const size_t buf_bytes  = buf_elems * sizeof(float);                      // 8 MB

    _Float16* xpad  = (_Float16*)ws;
    _Float16* wprep = (_Float16*)(ws + xpad_bytes);
    float* hbuf  = (float*)(ws + xpad_bytes + wprep_bytes);
    float* zbuf  = (float*)((char*)hbuf + buf_bytes);
    float* sbuf  = (float*)((char*)zbuf + buf_bytes);
    float* outws = (float*)((char*)sbuf + buf_bytes);

    const int npx = 2 * PIX_PAD * CIN;            // 2,515,456
    prep_x<<<(npx + 255) / 256, 256, 0, stream>>>(x, xpad, npx);
    prep_w<<<((int)wprep_n + 255) / 256, 256, 0, stream>>>(Wh, Wz, Ws, wprep, (int)wprep_n);
    zero_f32<<<((int)buf_elems + 255) / 256, 256, 0, stream>>>(outws, (int)buf_elems);

    for (int dir = 0; dir < 8; ++dir) {
        conv_wmma<<<512, 256, 0, stream>>>(xpad, wprep, bh, bz, bs, hbuf, zbuf, sbuf, dir);
        gru_scan<<<8192, 256, 0, stream>>>(zbuf, hbuf, sbuf, h0, outws, dir);
    }
    to_ncdhw<<<((int)buf_elems + 255) / 256, 256, 0, stream>>>(outws, out, (int)buf_elems);
}
